// SentenceToDocument_40484361732589
// MI455X (gfx1250) — matrix operand order repaired and checked by reading, not compiled, not run on hardware
//
#include <hip/hip_runtime.h>

typedef __attribute__((ext_vector_type(16))) __bf16 v16bf;
typedef __attribute__((ext_vector_type(2)))  __bf16 v2bf;
typedef __attribute__((ext_vector_type(16))) float  v16f;
typedef __attribute__((ext_vector_type(8)))  float  v8f;
typedef __attribute__((ext_vector_type(4)))  float  v4f;
typedef __attribute__((ext_vector_type(2)))  float  v2f;

#define B_BATCH 2048
#define S_TOT   801
#define D_DIM   100
#define U_DIM   50
#define NSENT   20
#define SEC     40          // sentences per section
#define ROWS    80          // rows handled per wave (2 sections => 5 exact M-tiles)

// True packed f32x2 -> bf16x2 convert (one v_cvt_pk_bf16_f32).
__device__ __forceinline__ unsigned int pk2bf(float lo, float hi) {
    v2f f; f.x = lo; f.y = hi;
    v2bf b = __builtin_convertvector(f, v2bf);
    return __builtin_bit_cast(unsigned int, b);
}
__device__ __forceinline__ float fast_rcp(float x) {
#if __has_builtin(__builtin_amdgcn_rcpf)
    return __builtin_amdgcn_rcpf(x);
#else
    return 1.0f / x;
#endif
}
__device__ __forceinline__ float fast_tanh(float x) {
    float e = __expf(2.0f * x);                 // inf-safe: +inf -> 1, 0 -> -1
    return 1.0f - 2.0f * fast_rcp(e + 1.0f);
}

// ---------------------------------------------------------------------------
// Kernel 1: pack wh (U x D fp32) into bf16 B-operand fragments, per-lane order.
// Fragment f = nt*4+ks; lane L supplies col = nt*16+(L&15);
// element j (0..15): K = ks*32 + (L>>4)*16 + j   (contiguous k order, same as A)
// ---------------------------------------------------------------------------
__global__ void pack_wh_kernel(const float* __restrict__ wh,
                               unsigned short* __restrict__ bpack) {
    int t = blockIdx.x * blockDim.x + threadIdx.x;   // 0..8191
    if (t >= 16 * 32 * 16) return;
    int j  = t & 15;
    int L  = (t >> 4) & 31;
    int f  = t >> 9;
    int nt = f >> 2, ks = f & 3;
    int col = nt * 16 + (L & 15);
    int K   = ks * 32 + ((L >> 4) << 4) + j;
    float v = (col < U_DIM && K < D_DIM) ? wh[col * D_DIM + K] : 0.0f;
    __bf16 bv = (__bf16)v;
    bpack[t] = __builtin_bit_cast(unsigned short, bv);
}

// ---------------------------------------------------------------------------
// Kernel 2: wub[b][u] = dot(uvec[b], wu[u]) + bw[u]
// ---------------------------------------------------------------------------
__global__ void wub_kernel(const float* __restrict__ inputs,
                           const float* __restrict__ wu,
                           const float* __restrict__ bw,
                           float* __restrict__ wub) {
    int b = blockIdx.x;
    int u = threadIdx.x;
    if (u >= U_DIM) return;
    const float* uv = inputs + ((size_t)b * S_TOT + (S_TOT - 1)) * D_DIM;
    const float* w  = wu + u * D_DIM;
    float acc = 0.f;
    #pragma unroll 4
    for (int d = 0; d < D_DIM; ++d) acc = fmaf(uv[d], w[d], acc);
    wub[b * U_DIM + u] = acc + bw[u];
}

// ---------------------------------------------------------------------------
// Kernel 3: main. One wave per 2 sections (80 rows x 100 cols).
// h staged once to LDS via GLOBAL_LOAD_ASYNC_TO_LDS_B128 (flat copy, natural
// row stride 100); bf16 WMMA GEMM (5 M-tiles x 4 N-tiles x 4 K-steps, last
// K-step masked to the 4 valid columns); tanh + vw dot + width-16 shuffle
// reduction -> e; softmax per 40; alpha-weighted row sum from LDS -> out.
// ---------------------------------------------------------------------------
__launch_bounds__(32)
__global__ void attn_main_kernel(const float* __restrict__ inputs,
                                 const float* __restrict__ vw,
                                 const unsigned short* __restrict__ bpack_us,
                                 const float* __restrict__ wub,
                                 float* __restrict__ out) {
    __shared__ float hs[ROWS * D_DIM + 32];   // +32 pad: benign tail reads
    __shared__ float e_s[ROWS];
    __shared__ float a_s[ROWS];

    const int L    = threadIdx.x;         // 0..31
    const int p    = blockIdx.x;          // 0..20479
    const int b    = p / (NSENT / 2);
    const int n0   = (p % (NSENT / 2)) * 2;
    const int rrow = L & 15;
    const int half = L >> 4;

    const float* hbase = inputs + ((size_t)b * S_TOT + n0 * SEC) * D_DIM;

    // ---- async DMA: 80*400B = 2000 x 16B chunks, global -> LDS, flat copy
    {
        unsigned lbase = (unsigned)(uintptr_t)(&hs[0]);
        for (int it = 0; it < 63; ++it) {
            int c = it * 32 + L;
            if (c < 2000) {
                unsigned off = (unsigned)c * 16u;
                asm volatile("global_load_async_to_lds_b128 %0, %1, %2"
                             :: "v"(lbase + off), "v"(off), "s"(hbase)
                             : "memory");
            }
        }
    }

    // ---- overlap: B fragments + per-lane bias/vw load while DMA runs
    const v16bf* bp = (const v16bf*)bpack_us;
    v16bf bfrag[4][4];
    #pragma unroll
    for (int nt = 0; nt < 4; ++nt)
        #pragma unroll
        for (int ks = 0; ks < 4; ++ks)
            bfrag[nt][ks] = bp[(nt * 4 + ks) * 32 + L];

    float wubv[4], vwv[4];
    #pragma unroll
    for (int nt = 0; nt < 4; ++nt) {
        int col  = nt * 16 + rrow;
        bool ok  = col < U_DIM;
        wubv[nt] = ok ? wub[b * U_DIM + col] : 0.0f;
        vwv[nt]  = ok ? vw[col] : 0.0f;
    }

    asm volatile("s_wait_asynccnt 0x0" ::: "memory");
    __syncthreads();

    for (int mt = 0; mt < 5; ++mt) {
        v8f acc[4];
        #pragma unroll
        for (int nt = 0; nt < 4; ++nt) acc[nt] = (v8f)(0.0f);

        const int r = mt * 16 + rrow;

        // K-steps 0..2 : full 32-wide bf16 WMMA steps (K 0..95)
        #pragma unroll
        for (int ks = 0; ks < 3; ++ks) {
            // A fragment (contiguous, matches B pack): K = ks*32 + half*16 + {0..15}
            int kb = ks * 32 + (half << 4);
            const v4f* rp = (const v4f*)(&hs[r * D_DIM + kb]);
            v4f f0 = rp[0], f1 = rp[1], f2 = rp[2], f3 = rp[3];
            v16f f;
            f[0]  = f0.x; f[1]  = f0.y; f[2]  = f0.z; f[3]  = f0.w;
            f[4]  = f1.x; f[5]  = f1.y; f[6]  = f1.z; f[7]  = f1.w;
            f[8]  = f2.x; f[9]  = f2.y; f[10] = f2.z; f[11] = f2.w;
            f[12] = f3.x; f[13] = f3.y; f[14] = f3.z; f[15] = f3.w;
            v16bf a = __builtin_convertvector(f, v16bf);   // 8x v_cvt_pk_bf16_f32
            #pragma unroll
            for (int nt = 0; nt < 4; ++nt) {
                acc[nt] = __builtin_amdgcn_wmma_f32_16x16x32_bf16(
                    false, a, false, bfrag[nt][ks], (short)0, acc[nt],
                    false, false);
            }
        }
        // K-step 3 : only K=96..99 are real (half-0, j=0..3); rest zero.
        {
            v4f ft = *(const v4f*)(&hs[r * D_DIM + 96]);   // same row both halves
            union { unsigned int u[8]; v16bf v; } au;
            bool lo = (half == 0);
            au.u[0] = lo ? pk2bf(ft.x, ft.y) : 0u;
            au.u[1] = lo ? pk2bf(ft.z, ft.w) : 0u;
            au.u[2] = 0u; au.u[3] = 0u;
            au.u[4] = 0u; au.u[5] = 0u; au.u[6] = 0u; au.u[7] = 0u;
            #pragma unroll
            for (int nt = 0; nt < 4; ++nt) {
                acc[nt] = __builtin_amdgcn_wmma_f32_16x16x32_bf16(
                    false, au.v, false, bfrag[nt][3], (short)0, acc[nt],
                    false, false);
            }
        }

        // epilogue: t = tanh(acc + wub), ev += t * vw  (per-lane partial cols)
        float ev[8];
        #pragma unroll
        for (int v = 0; v < 8; ++v) ev[v] = 0.f;
        #pragma unroll
        for (int nt = 0; nt < 4; ++nt) {
            #pragma unroll
            for (int v = 0; v < 8; ++v) {
                float t = fast_tanh(acc[nt][v] + wubv[nt]);
                ev[v] = fmaf(t, vwv[nt], ev[v]);
            }
        }
        // reduce over the 16 lanes of each half (cols) -> e per row
        #pragma unroll
        for (int v = 0; v < 8; ++v) {
            float s = ev[v];
            s += __shfl_xor(s, 1, 16);
            s += __shfl_xor(s, 2, 16);
            s += __shfl_xor(s, 4, 16);
            s += __shfl_xor(s, 8, 16);
            ev[v] = s;
        }
        if (rrow == 0) {
            #pragma unroll
            for (int v = 0; v < 8; ++v)
                e_s[mt * 16 + half * 8 + v] = ev[v];   // row = mt*16 + half*8 + v
        }
    }
    __syncthreads();

    // softmax over each 40-row section; half H owns section H
    {
        int base = half * SEC;
        float e0 = e_s[base + rrow];
        float e1 = e_s[base + 16 + rrow];
        float e2 = (rrow < 8) ? e_s[base + 32 + rrow] : -3.0e38f;
        float m  = fmaxf(e0, fmaxf(e1, e2));
        m = fmaxf(m, __shfl_xor(m, 1, 16));
        m = fmaxf(m, __shfl_xor(m, 2, 16));
        m = fmaxf(m, __shfl_xor(m, 4, 16));
        m = fmaxf(m, __shfl_xor(m, 8, 16));
        float x0 = __expf(e0 - m);
        float x1 = __expf(e1 - m);
        float x2 = (rrow < 8) ? __expf(e2 - m) : 0.f;
        float s  = x0 + x1 + x2;
        s += __shfl_xor(s, 1, 16);
        s += __shfl_xor(s, 2, 16);
        s += __shfl_xor(s, 4, 16);
        s += __shfl_xor(s, 8, 16);
        float inv = fast_rcp(s);
        a_s[base + rrow]      = x0 * inv;
        a_s[base + 16 + rrow] = x1 * inv;
        if (rrow < 8) a_s[base + 32 + rrow] = x2 * inv;
    }
    __syncthreads();

    // out[b, n0+sec, :] = sum_s alpha[s] * h[s, :]
    #pragma unroll
    for (int sec = 0; sec < 2; ++sec) {
        float o0 = 0.f, o1 = 0.f, o2 = 0.f, o3 = 0.f;
        for (int s = 0; s < SEC; ++s) {
            float a = a_s[sec * SEC + s];
            const float* row = &hs[(sec * SEC + s) * D_DIM];
            o0 = fmaf(a, row[L],       o0);
            o1 = fmaf(a, row[L + 32],  o1);
            o2 = fmaf(a, row[L + 64],  o2);
            o3 = fmaf(a, row[96 + L],  o3);   // only L<4 is kept
        }
        float* op = out + ((size_t)b * NSENT + n0 + sec) * D_DIM;
        op[L]      = o0;
        op[L + 32] = o1;
        op[L + 64] = o2;
        if (L < 4) op[96 + L] = o3;
    }
}

// ---------------------------------------------------------------------------
extern "C" void kernel_launch(void* const* d_in, const int* in_sizes, int n_in,
                              void* d_out, int out_size, void* d_ws, size_t ws_size,
                              hipStream_t stream) {
    (void)in_sizes; (void)n_in; (void)out_size; (void)ws_size;
    const float* inputs = (const float*)d_in[0];
    const float* vw     = (const float*)d_in[1];
    const float* wh     = (const float*)d_in[2];
    const float* wu     = (const float*)d_in[3];
    const float* bw     = (const float*)d_in[4];
    float* out = (float*)d_out;

    unsigned short* bpack = (unsigned short*)d_ws;              // 16 KiB
    float* wub = (float*)((char*)d_ws + 16 * 1024);             // 2048*50*4 = 400 KiB

    pack_wh_kernel<<<32, 256, 0, stream>>>(wh, bpack);
    wub_kernel<<<B_BATCH, 64, 0, stream>>>(inputs, wu, bw, wub);
    attn_main_kernel<<<B_BATCH * (NSENT / 2), 32, 0, stream>>>(
        inputs, vw, bpack, wub, out);
}
